// NTXentLoss_9680856285366
// MI455X (gfx1250) — matrix operand order repaired and checked by reading, not compiled, not run on hardware
//
#include <hip/hip_runtime.h>
#include <hip/hip_bf16.h>

#define B_ROWS 4096
#define NN 8192            // 2*B
#define DD 256             // feature dim
#define INVT 2.0f          // 1/temperature (T = 0.5)
#define LOG2E2 2.885390081777927f   // 2 * log2(e)  (exp(2x) == exp2(x*LOG2E2))
#define COL_BLOCKS 8
#define CPB (NN / COL_BLOCKS)   // 1024 columns per col-block

typedef __attribute__((ext_vector_type(16))) __bf16 v16bf;
typedef __attribute__((ext_vector_type(8)))  float  v8f;

// Raw hardware exp2 (v_exp_f32): our arguments are in [-2.9, 2.9], so the
// OCML denorm-range fixup (cmp/cndmask/scale per element) is pure overhead.
#if __has_builtin(__builtin_amdgcn_exp2f)
#define FAST_EXP2(x) __builtin_amdgcn_exp2f(x)
#else
#define FAST_EXP2(x) exp2f(x)
#endif

union Frag {
    v16bf v;
    unsigned u[8];
};

// ---------------------------------------------------------------------------
// Kernel 1: row-normalize z = [z_i ; z_j], round-to-nearest-even to bf16.
// One wave (32 lanes) per row; 8 rows per 256-thread block.
// ---------------------------------------------------------------------------
__global__ __launch_bounds__(256) void ntxent_normalize_kernel(
    const float* __restrict__ zi, const float* __restrict__ zj,
    unsigned short* __restrict__ zn /* bf16 bits, [NN][DD] */) {
    int lane = threadIdx.x & 31;
    int wave = threadIdx.x >> 5;
    int row  = blockIdx.x * 8 + wave;
    const float* src = (row < B_ROWS) ? (zi + (size_t)row * DD)
                                      : (zj + (size_t)(row - B_ROWS) * DD);
    float vals[8];
    float ss = 0.0f;
#pragma unroll
    for (int k = 0; k < 8; ++k) {
        vals[k] = src[lane + 32 * k];
        ss += vals[k] * vals[k];
    }
#pragma unroll
    for (int off = 1; off < 32; off <<= 1) ss += __shfl_xor(ss, off, 32);
    float nrm = fmaxf(sqrtf(ss), 1e-12f);
    float inv = 1.0f / nrm;
#pragma unroll
    for (int k = 0; k < 8; ++k) {
        float f = vals[k] * inv;
        unsigned bits = __float_as_uint(f);
        // round-to-nearest-even fp32 -> bf16
        unsigned r = (bits + 0x7FFFu + ((bits >> 16) & 1u)) >> 16;
        zn[(size_t)row * DD + lane + 32 * k] = (unsigned short)r;
    }
}

// ---------------------------------------------------------------------------
// Kernel 2: fused sim GEMM + exp-reduction.
// Grid: (NN/128 row blocks, COL_BLOCKS). Block: 256 threads = 8 waves.
// Each wave owns 16 rows; A panel (16x256 bf16) lives in 64 VGPRs.
// Per 16x16 column tile: 8x (b128-pair load clause + chained
// v_wmma_f32_16x16x32_bf16), then an unconditional raw-exp2 accumulate.
// Diagonal removal and positive-pair capture happen in scalar-guarded
// blocks that fire exactly once per wave.
// ---------------------------------------------------------------------------
__global__ __launch_bounds__(256) void ntxent_sim_kernel(
    const unsigned short* __restrict__ zn,
    float* __restrict__ rowsum /* [COL_BLOCKS][NN] */,
    float* __restrict__ rowpos /* [NN] */) {
    const unsigned* z32 = (const unsigned*)zn;  // dword view; DD/2 dwords per row
    int lane = threadIdx.x & 31;
    int wave = threadIdx.x >> 5;
    int ln   = lane & 15;       // position within 16-lane half
    int hi   = lane >> 4;       // 0 = lanes 0-15, 1 = lanes 16-31
    int rowbase  = blockIdx.x * 128 + wave * 16;
    int colblock = blockIdx.y;
    int colstart = colblock * CPB;
    int colend   = colstart + CPB;

    // Wave-uniform scalars: the (only) column tiles holding the diagonal and
    // the positive pair for this wave's 16 rows.
    int rb_s = __builtin_amdgcn_readfirstlane(rowbase);
    int pc_s = (rb_s < B_ROWS) ? (rb_s + B_ROWS) : (rb_s - B_ROWS);

    // --- Preload A fragments: 16-bit A 16x32 in contiguous k order (not the ISA table order) ---
    // lane (0-15 | 16-31) = row M; VGPR v: K=2v (+16 if hi-half),
    // matching B's contiguous order. Each dword = contiguous K pair.
    Frag a[8];
    {
        int arow = rowbase + ln;
        const unsigned* ap = z32 + (size_t)arow * (DD / 2);
#pragma unroll
        for (int ks = 0; ks < 8; ++ks) {
#pragma unroll
            for (int v = 0; v < 8; ++v) {
                int K = 32 * ks + 2 * v + (hi ? 16 : 0);
                a[ks].u[v] = ap[K >> 1];
            }
        }
    }

    float acc[8];
#pragma unroll
    for (int r = 0; r < 8; ++r) acc[r] = 0.0f;

    // B fragment: 16-bit B 32x16 layout — VGPR v holds K=2v,2v+1
    // (lanes 0-15, col=lane) / K=16+2v,+1 (lanes 16-31, col=lane-16).
    // B[k][nn] = zhat[ct+nn][k]: per lane the 8 dwords of fragment ks are
    // contiguous at dword index 16*ks (+8 for the hi half, folded into bp).
    // Loop-carried pointer: one 64-bit add per tile instead of a recompute.
    const unsigned* bp = z32 + (size_t)(colstart + ln) * (DD / 2) + 8 * hi;

    for (int ct = colstart; ct < colend; ct += 16, bp += 16 * (DD / 2)) {
        // Prefetch next tile's B rows while this tile's WMMA chain runs.
        __builtin_prefetch((const void*)(bp + 16 * (DD / 2)), 0, 1);
        Frag b[8];
#pragma unroll
        for (int ks = 0; ks < 8; ++ks) {
#pragma unroll
            for (int v = 0; v < 8; ++v) {
                b[ks].u[v] = bp[16 * ks + v];
            }
        }
        v8f c = {0.f, 0.f, 0.f, 0.f, 0.f, 0.f, 0.f, 0.f};
#pragma unroll
        for (int ks = 0; ks < 8; ++ks) {
            c = __builtin_amdgcn_wmma_f32_16x16x32_bf16(
                    false, a[ks].v, false, b[ks].v, (short)0, c, false, false);
        }
        // Unconditional epilogue: C/D layout — VGPR r: lanes 0-15 -> M=r,
        // lanes 16-31 -> M=r+8, N = lane%16. Raw v_exp_f32 (TRANS) co-executes
        // with the matrix pipe.
#pragma unroll
        for (int r = 0; r < 8; ++r) {
            acc[r] += FAST_EXP2(c[r] * LOG2E2);
        }
        // Diagonal tile: remove exp(sim_ii/T). Element (m,m) sits at lane
        // ln == r + 8*hi. Scalar-uniform branch; executes for one ct only.
        if (ct == rb_s) {
#pragma unroll
            for (int r = 0; r < 8; ++r) {
                if (ln == r + 8 * hi) acc[r] -= FAST_EXP2(c[r] * LOG2E2);
            }
        }
        // Positive-pair tile: sim[m, pair(m)] sits at lane ln == r + 8*hi.
        // Exactly one writer grid-wide per row. Executes for one ct only.
        if (ct == pc_s) {
#pragma unroll
            for (int r = 0; r < 8; ++r) {
                if (ln == r + 8 * hi) rowpos[rowbase + r + 8 * hi] = c[r];
            }
        }
    }

    // Reduce acc over the 16 lanes sharing each row (xor masks stay in-half).
#pragma unroll
    for (int r = 0; r < 8; ++r) {
        float s = acc[r];
#pragma unroll
        for (int off = 1; off < 16; off <<= 1) s += __shfl_xor(s, off, 32);
        if (ln == 0) rowsum[(size_t)colblock * NN + rowbase + r + hi * 8] = s;
    }
}

// ---------------------------------------------------------------------------
// Kernel 3: deterministic final reduction. One block of 256 threads.
// loss_i = -pos_i*INVT + log(sum_b rowsum[b][i]); out = mean over NN rows.
// ---------------------------------------------------------------------------
__global__ __launch_bounds__(256) void ntxent_loss_kernel(
    const float* __restrict__ rowsum, const float* __restrict__ rowpos,
    float* __restrict__ out) {
    __shared__ float red[256];
    int t = threadIdx.x;
    float s = 0.0f;
    for (int i = t; i < NN; i += 256) {
        float denom = 0.0f;
#pragma unroll
        for (int b = 0; b < COL_BLOCKS; ++b) denom += rowsum[(size_t)b * NN + i];
        s += logf(denom) - rowpos[i] * INVT;
    }
    red[t] = s;
    __syncthreads();
    for (int off = 128; off > 0; off >>= 1) {
        if (t < off) red[t] += red[t + off];
        __syncthreads();
    }
    if (t == 0) out[0] = red[0] * (1.0f / (float)NN);
}

extern "C" void kernel_launch(void* const* d_in, const int* in_sizes, int n_in,
                              void* d_out, int out_size, void* d_ws, size_t ws_size,
                              hipStream_t stream) {
    const float* zi = (const float*)d_in[0];
    const float* zj = (const float*)d_in[1];
    float* out = (float*)d_out;

    char* ws = (char*)d_ws;
    unsigned short* zn = (unsigned short*)ws;                       // 4 MB bf16
    float* rowsum = (float*)(ws + (size_t)NN * DD * 2);             // 256 KB
    float* rowpos = (float*)(ws + (size_t)NN * DD * 2
                                + (size_t)COL_BLOCKS * NN * 4);     // 32 KB

    ntxent_normalize_kernel<<<NN / 8, 256, 0, stream>>>(zi, zj, zn);
    ntxent_sim_kernel<<<dim3(NN / 128, COL_BLOCKS), 256, 0, stream>>>(zn, rowsum, rowpos);
    ntxent_loss_kernel<<<1, 256, 0, stream>>>(rowsum, rowpos, out);
}
